// AttentionBlock_29411936043499
// MI455X (gfx1250) — matrix operand order repaired and checked by reading, not compiled, not run on hardware
//
#include <hip/hip_runtime.h>
#include <hip/hip_bf16.h>
#include <math.h>

// ---------------------------------------------------------------------------
// Types for CDNA5 WMMA (wave32): v_wmma_f32_16x16x32_bf16
// ---------------------------------------------------------------------------
typedef __bf16 bf16_t;
typedef __attribute__((ext_vector_type(16))) __bf16 v16bf;
typedef __attribute__((ext_vector_type(8)))  __bf16 v8bf;
typedef __attribute__((ext_vector_type(2)))  __bf16 v2bf;
typedef __attribute__((ext_vector_type(8)))  float  v8f;
typedef __attribute__((ext_vector_type(4)))  int    v4i;

#define B_   8
#define C_   256
#define N_   4096
#define G_   8
#define CG_  32          // channels per group
#define EPS_ 1e-5f

// LDS tile strides (elements); all row starts 16B-aligned, bank-rotated
#define KT_LD 264        // 32 x 264 bf16  = 16896 B
#define VT_LD 40         // 256 x 40 bf16  = 20480 B
#define PT_LD 40         // 8 waves x 16 x 40 bf16 = 10240 B

#if __has_builtin(__builtin_amdgcn_global_load_async_to_lds_b128) && \
    __has_builtin(__builtin_amdgcn_s_wait_asynccnt)
#define ATTN_ASYNC 1
#else
#define ATTN_ASYNC 0
#endif

__device__ __forceinline__ v8f wmma_bf(v16bf a, v16bf b, v8f c) {
    return __builtin_amdgcn_wmma_f32_16x16x32_bf16(
        /*neg_a=*/false, a, /*neg_b=*/false, b,
        /*c_mod=*/(short)0, c, /*reuse_a=*/false, /*reuse_b=*/false);
}

// A-fragment (16 rows x 32 K, bf16), source row-major with leading dim `ld`.
// Layout: lanes 0-15 row M=lane hold K {0..15}; lanes 16-31 hold
// K {16..31}. Two 16-byte contiguous loads per lane.
__device__ __forceinline__ v16bf load_a_frag(const bf16_t* base, int ld,
                                             int row0, int k0) {
    int lane = threadIdx.x & 31;
    int row  = row0 + (lane & 15);
    int half = lane >> 4;
    const bf16_t* p = base + (size_t)row * ld + k0 + half * 16;
    v8bf lo = *(const v8bf*)(p);
    v8bf hi = *(const v8bf*)(p + 8);
    return __builtin_shufflevector(lo, hi, 0,1,2,3,4,5,6,7,8,9,10,11,12,13,14,15);
}

// B-fragment (32 K x 16 cols, bf16), source stored [col][k] contiguous in k.
// One 32-byte per-lane chunk (global).
__device__ __forceinline__ v16bf load_b_frag(const bf16_t* base, int ld,
                                             int col0, int k0) {
    int lane = threadIdx.x & 31;
    int col  = col0 + (lane & 15);
    int half = lane >> 4;
    return *(const v16bf*)(base + (size_t)col * ld + k0 + half * 16);
}

// B-fragment from LDS: two ds_load_b128 per lane.
__device__ __forceinline__ v16bf load_b_lds(const bf16_t* base, int ld,
                                            int col0, int k0) {
    int lane = threadIdx.x & 31;
    int col  = col0 + (lane & 15);
    int half = lane >> 4;
    const bf16_t* p = base + col * ld + k0 + half * 16;
    v8bf lo = *(const v8bf*)(p);
    v8bf hi = *(const v8bf*)(p + 8);
    return __builtin_shufflevector(lo, hi, 0,1,2,3,4,5,6,7,8,9,10,11,12,13,14,15);
}

// 16-byte global -> LDS copy: async DMA path on gfx1250, sync fallback else.
__device__ __forceinline__ void copy16_g2l(bf16_t* lds, const bf16_t* g) {
#if ATTN_ASYNC
    __builtin_amdgcn_global_load_async_to_lds_b128(
        (__attribute__((address_space(1))) v4i*)(g),
        (__attribute__((address_space(3))) v4i*)(lds), 0, 0);
#else
    *(v8bf*)lds = *(const v8bf*)g;
#endif
}
__device__ __forceinline__ void wait_stage() {
#if ATTN_ASYNC
    __builtin_amdgcn_s_wait_asynccnt(0);
#endif
}

// ---------------------------------------------------------------------------
// K1: GroupNorm statistics. One block per (b,g); reduce 32*4096 floats.
// ---------------------------------------------------------------------------
__global__ __launch_bounds__(256) void k_gn_stats(const float* __restrict__ x,
                                                  float* __restrict__ stats) {
    int bg = blockIdx.x;
    const float4* p4 = (const float4*)(x + (size_t)bg * (CG_ * N_));
    float s = 0.f, ss = 0.f;
    for (int i = threadIdx.x; i < (CG_ * N_) / 4; i += 256) {
        float4 v = p4[i];
        s  += v.x + v.y + v.z + v.w;
        ss += v.x*v.x + v.y*v.y + v.z*v.z + v.w*v.w;
    }
    for (int m = 1; m < 32; m <<= 1) {
        s  += __shfl_xor(s,  m, 32);
        ss += __shfl_xor(ss, m, 32);
    }
    __shared__ float aux[16];
    int w = threadIdx.x >> 5;
    if ((threadIdx.x & 31) == 0) { aux[w] = s; aux[8 + w] = ss; }
    __syncthreads();
    if (threadIdx.x == 0) {
        float S = 0.f, SS = 0.f;
        for (int i = 0; i < 8; ++i) { S += aux[i]; SS += aux[8 + i]; }
        const float inv = 1.0f / (float)(CG_ * N_);
        float mean = S * inv;
        float var  = SS * inv - mean * mean;
        stats[bg * 2]     = mean;
        stats[bg * 2 + 1] = rsqrtf(var + EPS_);
    }
}

// ---------------------------------------------------------------------------
// K2: normalize + affine, transpose (B,C,N) -> (B,N,C), cast to bf16.
// ---------------------------------------------------------------------------
__global__ __launch_bounds__(256) void k_norm_t(const float* __restrict__ x,
                                                const float* __restrict__ stats,
                                                const float* __restrict__ sc,
                                                const float* __restrict__ bi,
                                                bf16_t* __restrict__ xnT) {
    int id = blockIdx.x;
    int nt = id & 63;
    int ct = (id >> 6) & 3;
    int b  = id >> 8;
    int t  = threadIdx.x;
    __shared__ __attribute__((aligned(16))) bf16_t sm[64][65];
    int n0 = nt * 64, c0 = ct * 64;
    int nl = t & 63;
    int cb = t >> 6;
    for (int r = 0; r < 16; ++r) {
        int cl = cb + r * 4;
        int c  = c0 + cl;
        float v = x[((size_t)b * C_ + c) * N_ + n0 + nl];
        int g = c >> 5;
        float mean = stats[(b * G_ + g) * 2];
        float rs   = stats[(b * G_ + g) * 2 + 1];
        v = (v - mean) * rs * sc[c] + bi[c];
        sm[cl][nl] = (bf16_t)v;
    }
    __syncthreads();
    int cp = t & 31;
    int nb = t >> 5;
    unsigned* out = (unsigned*)xnT;
    for (int r = 0; r < 8; ++r) {
        int nl2 = nb + r * 8;
        v2bf pk = { sm[cp * 2][nl2], sm[cp * 2 + 1][nl2] };
        out[(((size_t)b * N_ + n0 + nl2) * C_ + c0) / 2 + cp] =
            __builtin_bit_cast(unsigned, pk);
    }
}

// ---------------------------------------------------------------------------
// K3: cast weights to bf16
// ---------------------------------------------------------------------------
__global__ __launch_bounds__(256) void k_cast_w(const float* __restrict__ wq,
                                                const float* __restrict__ wo,
                                                bf16_t* __restrict__ wqb,
                                                bf16_t* __restrict__ wob) {
    int i = blockIdx.x * 256 + threadIdx.x;
    if (i < 3 * C_ * C_) wqb[i] = (bf16_t)wq[i];
    if (i < C_ * C_)     wob[i] = (bf16_t)wo[i];
}

// ---------------------------------------------------------------------------
// K4: QKV GEMM. Q stored (n,c) pre-scaled by 1/sqrt(C); K (n,c); V (c,n).
// ---------------------------------------------------------------------------
__global__ __launch_bounds__(256) void k_qkv(const bf16_t* __restrict__ xnT,
                                             const bf16_t* __restrict__ wq,
                                             const float*  __restrict__ bq,
                                             bf16_t* __restrict__ q,
                                             bf16_t* __restrict__ k,
                                             bf16_t* __restrict__ vT) {
    int b    = blockIdx.z;
    int wid  = threadIdx.x >> 5;
    int lane = threadIdx.x & 31;
    int nb   = blockIdx.x * 128 + (wid & 3) * 32;
    int ob   = blockIdx.y * 64  + (wid >> 2) * 32;
    const bf16_t* A = xnT + (size_t)b * N_ * C_;

    v8f acc[2][2] = {};
    for (int kk = 0; kk < C_; kk += 32) {
        v16bf a0 = load_a_frag(A, C_, nb,      kk);
        v16bf a1 = load_a_frag(A, C_, nb + 16, kk);
        v16bf b0 = load_b_frag(wq, C_, ob,      kk);
        v16bf b1 = load_b_frag(wq, C_, ob + 16, kk);
        acc[0][0] = wmma_bf(a0, b0, acc[0][0]);
        acc[0][1] = wmma_bf(a0, b1, acc[0][1]);
        acc[1][0] = wmma_bf(a1, b0, acc[1][0]);
        acc[1][1] = wmma_bf(a1, b1, acc[1][1]);
    }

    int half = lane >> 4, cl = lane & 15;
    const float scale = 0.0625f;                  // 1/sqrt(256)
    for (int ti = 0; ti < 2; ++ti)
        for (int tj = 0; tj < 2; ++tj) {
            int o    = ob + tj * 16 + cl;
            float bias = bq[o];
            int rowb = nb + ti * 16 + half * 8;
            if (o < C_) {                          // Q (scaled)
                for (int r = 0; r < 8; ++r)
                    q[((size_t)b * N_ + rowb + r) * C_ + o] =
                        (bf16_t)((acc[ti][tj][r] + bias) * scale);
            } else if (o < 2 * C_) {               // K
                int oc = o - C_;
                for (int r = 0; r < 8; ++r)
                    k[((size_t)b * N_ + rowb + r) * C_ + oc] =
                        (bf16_t)(acc[ti][tj][r] + bias);
            } else {                               // V transposed (c,n)
                int oc = o - 2 * C_;
                v8bf pk;
                for (int r = 0; r < 8; ++r)
                    pk[r] = (bf16_t)(acc[ti][tj][r] + bias);
                *(v8bf*)(vT + ((size_t)b * C_ + oc) * N_ + rowb) = pk;
            }
        }
}

// ---------------------------------------------------------------------------
// K5: flash attention with async global->LDS staging of K/V tiles.
// Block = 8 waves, 128 query rows; K/V streamed in 32-column tiles shared by
// all waves through LDS (ASYNCcnt path on gfx1250). grid = (N/128, B).
// ---------------------------------------------------------------------------
__global__ __launch_bounds__(256) void k_attn(const bf16_t* __restrict__ q,
                                              const bf16_t* __restrict__ kmat,
                                              const bf16_t* __restrict__ vT,
                                              bf16_t* __restrict__ attn) {
    int b    = blockIdx.y;
    int wid  = threadIdx.x >> 5;
    int lane = threadIdx.x & 31;
    int t    = threadIdx.x;
    int nw   = blockIdx.x * 128 + wid * 16;       // first query row of wave
    int half = lane >> 4, cl = lane & 15;

    const bf16_t* qb = q    + (size_t)b * N_ * C_;
    const bf16_t* kb = kmat + (size_t)b * N_ * C_;
    const bf16_t* vb = vT   + (size_t)b * C_ * N_;

    __shared__ __attribute__((aligned(32))) bf16_t Kt[32 * KT_LD];   // 16896 B
    __shared__ __attribute__((aligned(32))) bf16_t Vt[256 * VT_LD];  // 20480 B
    __shared__ __attribute__((aligned(32))) bf16_t Pt[8][16 * PT_LD];// 10240 B
    bf16_t* myp = Pt[wid];

    v8f acc[16] = {};
    float M[8], L[8];
    for (int r = 0; r < 8; ++r) { M[r] = -INFINITY; L[r] = 0.f; }

    for (int m0 = 0; m0 < N_; m0 += 32) {
        // ---- stage K(32x256) + V(256x32) tiles into LDS (2048 x 16B) ----
        for (int i = 0; i < 8; ++i) {
            int chunk = i * 256 + t;
            if (chunk < 1024) {                     // K tile
                int row = chunk >> 5, col = chunk & 31;
                copy16_g2l(Kt + row * KT_LD + col * 8,
                           kb + (size_t)(m0 + row) * C_ + col * 8);
            } else {                                // V tile
                int vi = chunk - 1024;
                int c = vi >> 2, cc = vi & 3;
                copy16_g2l(Vt + c * VT_LD + cc * 8,
                           vb + (size_t)c * N_ + m0 + cc * 8);
            }
        }
        if (m0 + 32 < N_)
            __builtin_prefetch(kb + (size_t)(m0 + 32) * C_, 0, 0);
        wait_stage();
        __syncthreads();                            // tiles visible

        // ----- S = Q K^T (16 x 32 tile as two 16x16 D fragments) -----
        v8f s0 = {}, s1 = {};
        for (int kk = 0; kk < C_; kk += 32) {
            v16bf aq = load_a_frag(qb, C_, nw, kk);
            v16bf b0 = load_b_lds(Kt, KT_LD, 0,  kk);
            v16bf b1 = load_b_lds(Kt, KT_LD, 16, kk);
            s0 = wmma_bf(aq, b0, s0);
            s1 = wmma_bf(aq, b1, s1);
        }

        // ----- online softmax over the 32 columns -----
        float alpha[8];
        for (int r = 0; r < 8; ++r) {
            float mx = fmaxf(s0[r], s1[r]);
            for (int d = 1; d < 16; d <<= 1) mx = fmaxf(mx, __shfl_xor(mx, d, 32));
            float nm = fmaxf(M[r], mx);
            alpha[r] = __expf(M[r] - nm);
            M[r] = nm;
            float p0 = __expf(s0[r] - nm);
            float p1 = __expf(s1[r] - nm);
            s0[r] = p0; s1[r] = p1;
            float rs = p0 + p1;
            for (int d = 1; d < 16; d <<= 1) rs += __shfl_xor(rs, d, 32);
            L[r] = L[r] * alpha[r] + rs;
        }
        for (int ct = 0; ct < 16; ++ct)
            for (int r = 0; r < 8; ++r) acc[ct][r] *= alpha[r];

        // ----- P -> LDS (D layout -> A-source tile) -----
        for (int r = 0; r < 8; ++r) {
            int row = r + half * 8;
            myp[row * PT_LD + cl]      = (bf16_t)s0[r];
            myp[row * PT_LD + 16 + cl] = (bf16_t)s1[r];
        }
        __syncthreads();                            // P visible

        // ----- P A-fragment from LDS -----
        v16bf ap;
        {
            const bf16_t* pp = myp + cl * PT_LD + half * 16;
            v8bf lo = *(const v8bf*)(pp);
            v8bf hi = *(const v8bf*)(pp + 8);
            ap = __builtin_shufflevector(lo, hi,
                     0,1,2,3,4,5,6,7,8,9,10,11,12,13,14,15);
        }

        // ----- O += P * V (16 column tiles of C) -----
        for (int ct = 0; ct < 16; ++ct) {
            v16bf bv = load_b_lds(Vt, VT_LD, ct * 16, 0);
            acc[ct] = wmma_bf(ap, bv, acc[ct]);
        }
        __syncthreads();            // all tile reads done before next overwrite
    }

    // normalize by row sums and store (n,c) bf16
    for (int ct = 0; ct < 16; ++ct)
        for (int r = 0; r < 8; ++r) {
            float o = acc[ct][r] / L[r];
            attn[((size_t)b * N_ + nw + half * 8 + r) * C_ + ct * 16 + cl] =
                (bf16_t)o;
        }
}

// ---------------------------------------------------------------------------
// K6: output projection + bias + residual, output (B,C,N) fp32.
// ---------------------------------------------------------------------------
__global__ __launch_bounds__(256) void k_proj(const bf16_t* __restrict__ attn,
                                              const bf16_t* __restrict__ wo,
                                              const float*  __restrict__ bo,
                                              const float*  __restrict__ x,
                                              float* __restrict__ out) {
    int b    = blockIdx.z;
    int wid  = threadIdx.x >> 5;
    int lane = threadIdx.x & 31;
    int nb   = blockIdx.x * 128 + (wid & 3) * 32;
    int ob   = blockIdx.y * 64  + (wid >> 2) * 32;
    const bf16_t* A = attn + (size_t)b * N_ * C_;

    v8f acc[2][2] = {};
    for (int kk = 0; kk < C_; kk += 32) {
        v16bf a0 = load_a_frag(A, C_, nb,      kk);
        v16bf a1 = load_a_frag(A, C_, nb + 16, kk);
        v16bf b0 = load_b_frag(wo, C_, ob,      kk);
        v16bf b1 = load_b_frag(wo, C_, ob + 16, kk);
        acc[0][0] = wmma_bf(a0, b0, acc[0][0]);
        acc[0][1] = wmma_bf(a0, b1, acc[0][1]);
        acc[1][0] = wmma_bf(a1, b0, acc[1][0]);
        acc[1][1] = wmma_bf(a1, b1, acc[1][1]);
    }

    int half = lane >> 4, cl = lane & 15;
    for (int ti = 0; ti < 2; ++ti)
        for (int tj = 0; tj < 2; ++tj) {
            int o    = ob + tj * 16 + cl;
            float bias = bo[o];
            int rowb = nb + ti * 16 + half * 8;
            const float* xr  = x   + ((size_t)b * C_ + o) * N_ + rowb;
            float*      orow = out + ((size_t)b * C_ + o) * N_ + rowb;
            for (int r = 0; r < 8; ++r)
                orow[r] = acc[ti][tj][r] + bias + xr[r];
        }
}

// ---------------------------------------------------------------------------
// Workspace layout (bytes)
// ---------------------------------------------------------------------------
static constexpr size_t O_STATS = 0;
static constexpr size_t O_XNT   = 1024;
static constexpr size_t O_WQKV  = O_XNT  + (size_t)B_ * N_ * C_ * 2;
static constexpr size_t O_WOUT  = O_WQKV + (size_t)3 * C_ * C_ * 2;
static constexpr size_t O_Q     = O_WOUT + (size_t)C_ * C_ * 2;
static constexpr size_t O_K     = O_Q    + (size_t)B_ * N_ * C_ * 2;
static constexpr size_t O_VT    = O_K    + (size_t)B_ * N_ * C_ * 2;
static constexpr size_t O_ATTN  = O_VT   + (size_t)B_ * N_ * C_ * 2;

extern "C" void kernel_launch(void* const* d_in, const int* in_sizes, int n_in,
                              void* d_out, int out_size, void* d_ws, size_t ws_size,
                              hipStream_t stream) {
    const float* x        = (const float*)d_in[0];
    const float* gn_scale = (const float*)d_in[1];
    const float* gn_bias  = (const float*)d_in[2];
    const float* w_qkv    = (const float*)d_in[3];
    const float* b_qkv    = (const float*)d_in[4];
    const float* w_out    = (const float*)d_in[5];
    const float* b_out    = (const float*)d_in[6];
    float* out = (float*)d_out;

    char* ws = (char*)d_ws;
    float*  stats = (float*) (ws + O_STATS);
    bf16_t* xnT   = (bf16_t*)(ws + O_XNT);
    bf16_t* wqb   = (bf16_t*)(ws + O_WQKV);
    bf16_t* wob   = (bf16_t*)(ws + O_WOUT);
    bf16_t* qbuf  = (bf16_t*)(ws + O_Q);
    bf16_t* kbuf  = (bf16_t*)(ws + O_K);
    bf16_t* vTbuf = (bf16_t*)(ws + O_VT);
    bf16_t* attn  = (bf16_t*)(ws + O_ATTN);

    k_gn_stats<<<B_ * G_, 256, 0, stream>>>(x, stats);
    k_norm_t<<<B_ * 4 * 64, 256, 0, stream>>>(x, stats, gn_scale, gn_bias, xnT);
    k_cast_w<<<(3 * C_ * C_) / 256, 256, 0, stream>>>(w_qkv, w_out, wqb, wob);
    k_qkv<<<dim3(N_ / 128, (3 * C_) / 64, B_), 256, 0, stream>>>(
        xnT, wqb, b_qkv, qbuf, kbuf, vTbuf);
    k_attn<<<dim3(N_ / 128, B_, 1), 256, 0, stream>>>(qbuf, kbuf, vTbuf, attn);
    k_proj<<<dim3(N_ / 128, C_ / 64, B_), 256, 0, stream>>>(
        attn, wob, b_out, x, out);
}
